// MultiHeadSelfAttention_35545149342346
// MI455X (gfx1250) — matrix operand order repaired and checked by reading, not compiled, not run on hardware
//
#include <hip/hip_runtime.h>

// MI455X (gfx1250) multi-head self-attention, wave32 + v_wmma_f32_16x16x32_f16.
// fp32 inputs downconverted to f16; all GEMMs accumulate fp32 via WMMA.
// Flash attention (online softmax); scores never hit HBM.
// Register blocking: 32x64 per wave in GEMMs, 32 queries per wave in attention.

#define BATCH 4
#define SEQ   2048
#define EMBED 1024
#define NH    16
#define HD    64

typedef __attribute__((ext_vector_type(16))) _Float16 v16h;
typedef __attribute__((ext_vector_type(8)))  _Float16 v8h;
typedef __attribute__((ext_vector_type(8)))  float    v8f;

static __device__ __forceinline__ v8f wmma16(v16h a, v16h b, v8f c) {
  // 8 args: (neg_a, A, neg_b, B, c_mod, C, reuse_a, reuse_b)
  return __builtin_amdgcn_wmma_f32_16x16x32_f16(false, a, false, b, (short)0, c,
                                                false, false);
}

// A-fragment 16x32 (MxK) f16: lane L holds row M=L%16.
// halves 0..7  = K in [8*hi, 8*hi+8); halves 8..15 = K in [16+8*hi, ...)
static __device__ __forceinline__ v16h load_a_frag(const _Float16* row_k0, int hi) {
  union { v16h v; v8h h[2]; } u;
  u.h[0] = *(const v8h*)(row_k0 + 8 * hi);
  u.h[1] = *(const v8h*)(row_k0 + 16 + 8 * hi);
  return u.v;
}

// B-fragment 32x16 (KxN) f16: lane L holds column N=L%16.
// halves 0..7  = K in [8*hi, 8*hi+8); halves 8..15 = K in [16+8*hi, ...)
static __device__ __forceinline__ v16h load_b_frag(const _Float16* col_k0, int hi) {
  union { v16h v; v8h h[2]; } u;
  u.h[0] = *(const v8h*)(col_k0 + 8 * hi);
  u.h[1] = *(const v8h*)(col_k0 + 16 + 8 * hi);
  return u.v;
}

// ---------------------------------------------------------------- conversions
__global__ void k_cvt_f32_f16(const float* __restrict__ src,
                              _Float16* __restrict__ dst, int n) {
  int i = blockIdx.x * blockDim.x + threadIdx.x;
  if (i < n) dst[i] = (_Float16)src[i];
}

// ---------------------------------------------------------------- QKV GEMM
// C[8192,3072] = xb[8192,1024] @ Wq[3072,1024]^T + bias, scattered into
// Q [B,H,S,Dh], K [B,H,S,Dh], Vt [B,H,Dh,S] (f16).
// Wave tile 32x64; the 8 waves of a block share the same 64 B-columns.
__global__ __launch_bounds__(256)
void k_qkv_gemm(const _Float16* __restrict__ xb, const _Float16* __restrict__ Wqb,
                const float* __restrict__ b_qkv,
                _Float16* __restrict__ Qb, _Float16* __restrict__ Kb,
                _Float16* __restrict__ Vtb) {
  const int K  = EMBED;
  const int TM = (BATCH * SEQ) / 32;               // 256 M-tiles
  int w    = blockIdx.x * 8 + (threadIdx.x >> 5);
  int lane = threadIdx.x & 31;
  int tm = w % TM, tn = w / TM;                    // block shares tn (8 | 256)
  int m0 = tm * 32, n0 = tn * 64;
  int nm = lane & 15, hi = lane >> 4;

  v8f acc[2][4] = {};
  const _Float16* arow0 = xb + (size_t)(m0 + nm) * K;
  const _Float16* arow1 = xb + (size_t)(m0 + 16 + nm) * K;

  for (int k0 = 0; k0 < K; k0 += 32) {
    v16h a0 = load_a_frag(arow0 + k0, hi);
    v16h a1 = load_a_frag(arow1 + k0, hi);
#pragma unroll
    for (int t = 0; t < 4; ++t) {
      v16h b = load_b_frag(Wqb + (size_t)(n0 + 16 * t + nm) * K + k0, hi);
      acc[0][t] = wmma16(a0, b, acc[0][t]);
      acc[1][t] = wmma16(a1, b, acc[1][t]);
    }
  }

#pragma unroll
  for (int t = 0; t < 4; ++t) {
    int ng   = n0 + 16 * t + nm;                   // global output column
    float bs = b_qkv[ng];
    int part = ng >> 10;                           // 0=Q 1=K 2=V (wave-uniform)
    int d    = ng & 1023;
    int h    = d >> 6, dh = d & 63;
#pragma unroll
    for (int mt = 0; mt < 2; ++mt)
#pragma unroll
      for (int r = 0; r < 8; ++r) {
        int row = m0 + 16 * mt + r + 8 * hi;       // global row (b,s)
        int b = row >> 11, s = row & 2047;
        _Float16 hv = (_Float16)(acc[mt][t][r] + bs);
        if (part == 0)
          Qb[((size_t)(b * NH + h) * SEQ + s) * HD + dh] = hv;
        else if (part == 1)
          Kb[((size_t)(b * NH + h) * SEQ + s) * HD + dh] = hv;
        else
          Vtb[((size_t)(b * NH + h) * HD + dh) * SEQ + s] = hv;
      }
  }
}

// ---------------------------------------------------------------- attention
// One wave: (b,h) head, 32 queries (two 16-row M-tiles), flash attention over
// all 2048 keys, 32 keys per iteration. K/V fragments reused by both M-tiles.
__global__ __launch_bounds__(256)
void k_attn(const _Float16* __restrict__ Qb, const _Float16* __restrict__ Kb,
            const _Float16* __restrict__ Vtb, _Float16* __restrict__ Ob) {
  __shared__ _Float16 smP[8][32 * 32];             // 2KB/wave P staging
  int wl   = threadIdx.x >> 5;
  int lane = threadIdx.x & 31;
  int w    = blockIdx.x * 8 + wl;
  _Float16* sp = smP[wl];

  const int QT = SEQ / 32;                         // 64 query tiles per head
  int bh = w / QT, qt = w % QT;                    // block shares bh (8 | 64)
  int q0 = qt * 32;
  const _Float16* Qp = Qb  + (size_t)bh * SEQ * HD;
  const _Float16* Kp = Kb  + (size_t)bh * SEQ * HD;
  const _Float16* Vp = Vtb + (size_t)bh * HD * SEQ;
  int nm = lane & 15, hi = lane >> 4;

  // Q tiles as A fragments: Dh=64 -> 2 K-steps of 32, two M-tiles
  v16h aq[2][2];
#pragma unroll
  for (int mt = 0; mt < 2; ++mt)
#pragma unroll
    for (int kf = 0; kf < 2; ++kf)
      aq[mt][kf] =
          load_a_frag(Qp + (size_t)(q0 + 16 * mt + nm) * HD + kf * 32, hi);

  v8f acc[2][4] = {};
  float m_i[2][8], l_i[2][8];
#pragma unroll
  for (int mt = 0; mt < 2; ++mt)
#pragma unroll
    for (int r = 0; r < 8; ++r) { m_i[mt][r] = -1e30f; l_i[mt][r] = 0.0f; }

  const float scale = 0.125f;                      // 1/sqrt(64)

  for (int kb = 0; kb < SEQ; kb += 32) {
    // --- K fragments, loaded once, reused by both M-tiles
    v16h bk[2][2];
#pragma unroll
    for (int sub = 0; sub < 2; ++sub)
#pragma unroll
      for (int kf = 0; kf < 2; ++kf)
        bk[sub][kf] =
            load_b_frag(Kp + (size_t)(kb + 16 * sub + nm) * HD + kf * 32, hi);

#pragma unroll
    for (int mt = 0; mt < 2; ++mt) {
      v8f s01[2];
#pragma unroll
      for (int sub = 0; sub < 2; ++sub) {
        v8f s = {};
        s = wmma16(aq[mt][0], bk[sub][0], s);
        s = wmma16(aq[mt][1], bk[sub][1], s);
        s01[sub] = s * scale;
      }
      // --- online softmax (rows live across 16-lane halves)
#pragma unroll
      for (int r = 0; r < 8; ++r) {
        float v0 = s01[0][r], v1 = s01[1][r];
        float mx = fmaxf(v0, v1);
#pragma unroll
        for (int off = 8; off >= 1; off >>= 1)
          mx = fmaxf(mx, __shfl_xor(mx, off, 32));
        float newm = fmaxf(m_i[mt][r], mx);
        float p0 = __expf(v0 - newm);
        float p1 = __expf(v1 - newm);
        float corr = __expf(m_i[mt][r] - newm);
        m_i[mt][r] = newm;
        float rs = p0 + p1;
#pragma unroll
        for (int off = 8; off >= 1; off >>= 1)
          rs += __shfl_xor(rs, off, 32);
        l_i[mt][r] = l_i[mt][r] * corr + rs;
        acc[mt][0][r] *= corr; acc[mt][1][r] *= corr;
        acc[mt][2][r] *= corr; acc[mt][3][r] *= corr;
        // stage P (f16) row-major [32 rows][32 keys] in LDS
        int rowi = 16 * mt + r + 8 * hi;
        sp[rowi * 32 + nm]      = (_Float16)p0;
        sp[rowi * 32 + 16 + nm] = (_Float16)p1;
      }
    }
    asm volatile("s_wait_dscnt 0" ::: "memory");   // C-layout -> A-layout handoff

    // --- P @ V : P as A-frags (from LDS), V^T rows as B-frags (reused x2)
    v16h ap[2];
#pragma unroll
    for (int mt = 0; mt < 2; ++mt) {
      union { v16h v; v8h h[2]; } u;
      const _Float16* prow = sp + (16 * mt + nm) * 32;
      u.h[0] = *(const v8h*)(prow + 8 * hi);
      u.h[1] = *(const v8h*)(prow + 16 + 8 * hi);
      ap[mt] = u.v;
    }
#pragma unroll
    for (int t = 0; t < 4; ++t) {
      v16h bv = load_b_frag(Vp + (size_t)(16 * t + nm) * SEQ + kb, hi);
      acc[0][t] = wmma16(ap[0], bv, acc[0][t]);
      acc[1][t] = wmma16(ap[1], bv, acc[1][t]);
    }
  }

  // --- normalize, write O [B, S, H*Dh] f16
  int b = bh >> 4, h = bh & 15;
#pragma unroll
  for (int mt = 0; mt < 2; ++mt)
#pragma unroll
    for (int t = 0; t < 4; ++t)
#pragma unroll
      for (int r = 0; r < 8; ++r) {
        int row = q0 + 16 * mt + r + 8 * hi;
        float val = acc[mt][t][r] / l_i[mt][r];
        Ob[((size_t)(b * SEQ + row)) * EMBED + h * HD + 16 * t + nm] =
            (_Float16)val;
      }
}

// ---------------------------------------------------------------- out proj
// out[8192,1024] = Ob[8192,1024] @ W_out[1024,1024]^T + b_out (fp32 store)
__global__ __launch_bounds__(256)
void k_out_gemm(const _Float16* __restrict__ Ob, const _Float16* __restrict__ Wob,
                const float* __restrict__ b_out, float* __restrict__ out) {
  const int K  = EMBED;
  const int TM = (BATCH * SEQ) / 32;               // 256 M-tiles
  int w    = blockIdx.x * 8 + (threadIdx.x >> 5);
  int lane = threadIdx.x & 31;
  int tm = w % TM, tn = w / TM;                    // block shares tn
  int m0 = tm * 32, n0 = tn * 64;
  int nm = lane & 15, hi = lane >> 4;

  v8f acc[2][4] = {};
  const _Float16* arow0 = Ob + (size_t)(m0 + nm) * K;
  const _Float16* arow1 = Ob + (size_t)(m0 + 16 + nm) * K;

  for (int k0 = 0; k0 < K; k0 += 32) {
    v16h a0 = load_a_frag(arow0 + k0, hi);
    v16h a1 = load_a_frag(arow1 + k0, hi);
#pragma unroll
    for (int t = 0; t < 4; ++t) {
      v16h b = load_b_frag(Wob + (size_t)(n0 + 16 * t + nm) * K + k0, hi);
      acc[0][t] = wmma16(a0, b, acc[0][t]);
      acc[1][t] = wmma16(a1, b, acc[1][t]);
    }
  }

#pragma unroll
  for (int t = 0; t < 4; ++t) {
    int col  = n0 + 16 * t + nm;
    float bs = b_out[col];
#pragma unroll
    for (int mt = 0; mt < 2; ++mt)
#pragma unroll
      for (int r = 0; r < 8; ++r) {
        int row = m0 + 16 * mt + r + 8 * hi;
        out[(size_t)row * EMBED + col] = acc[mt][t][r] + bs;
      }
  }
}

// ---------------------------------------------------------------- launcher
extern "C" void kernel_launch(void* const* d_in, const int* in_sizes, int n_in,
                              void* d_out, int out_size, void* d_ws, size_t ws_size,
                              hipStream_t stream) {
  const float* x     = (const float*)d_in[0];
  const float* W_qkv = (const float*)d_in[1];
  const float* b_qkv = (const float*)d_in[2];
  const float* W_out = (const float*)d_in[3];
  const float* b_out = (const float*)d_in[4];
  float* out = (float*)d_out;

  const size_t nX  = (size_t)BATCH * SEQ * EMBED;  // 8,388,608
  const size_t nWq = (size_t)3 * EMBED * EMBED;    // 3,145,728
  const size_t nWo = (size_t)EMBED * EMBED;        // 1,048,576

  _Float16* ws  = (_Float16*)d_ws;
  _Float16* xb  = ws;            size_t off = nX;
  _Float16* Wqb = ws + off;      off += nWq;
  _Float16* Wob = ws + off;      off += nWo;
  _Float16* Qb  = ws + off;      off += nX;
  _Float16* Kb  = ws + off;      off += nX;
  _Float16* Vtb = ws + off;      off += nX;
  _Float16* Ob  = ws + off;      off += nX;        // total ~92.3 MB

  k_cvt_f32_f16<<<(int)((nX  + 255) / 256), 256, 0, stream>>>(x,     xb,  (int)nX);
  k_cvt_f32_f16<<<(int)((nWq + 255) / 256), 256, 0, stream>>>(W_qkv, Wqb, (int)nWq);
  k_cvt_f32_f16<<<(int)((nWo + 255) / 256), 256, 0, stream>>>(W_out, Wob, (int)nWo);

  // (8192/32)*(3072/64) = 12288 waves -> 1536 blocks of 8 waves
  k_qkv_gemm<<<1536, 256, 0, stream>>>(xb, Wqb, b_qkv, Qb, Kb, Vtb);

  // 4*16*(2048/32) = 4096 waves -> 512 blocks
  k_attn<<<512, 256, 0, stream>>>(Qb, Kb, Vtb, Ob);

  // (8192/32)*(1024/64) = 4096 waves -> 512 blocks
  k_out_gemm<<<512, 256, 0, stream>>>(Ob, Wob, b_out, out);
}
